// SelfAttention_1202590843560
// MI455X (gfx1250) — matrix operand order repaired and checked by reading, not compiled, not run on hardware
//
#include <hip/hip_runtime.h>
#include <hip/hip_bf16.h>
#include <hip/hip_fp16.h>

// MI455X / gfx1250: wave32, WMMA 16x16x32 f16 -> f32, async global->LDS copies
typedef __attribute__((ext_vector_type(16))) _Float16    v16h;
typedef __attribute__((ext_vector_type(8)))  float       v8f;
typedef __attribute__((ext_vector_type(4)))  unsigned int uint4v;

#define BATCH   8
#define SEQ     2048
#define DMODEL  1024

#define EPI_F32            0
#define EPI_BIAS_F16       1
#define EPI_BIAS_F16_TRANS 2

// CDNA5 async direct global->LDS copy (ASYNCcnt-tracked); no clang builtin, so inline asm.
// GV mode: 64-bit global address in a VGPR pair, LDS byte address in a VGPR.
__device__ __forceinline__ void async_copy_b128(unsigned lds_addr, unsigned long long gaddr) {
    asm volatile("global_load_async_to_lds_b128 %0, %1, off"
                 :: "v"(lds_addr), "v"(gaddr) : "memory");
}
__device__ __forceinline__ void wait_async0() {
    asm volatile("s_wait_asynccnt 0x0" ::: "memory");
}

// ---------------- cast x (f32 -> f16), 8 elems / thread ----------------
__global__ void __launch_bounds__(256)
cast_f32_to_f16(const float* __restrict__ in, _Float16* __restrict__ out, int n) {
    int i = (blockIdx.x * 256 + threadIdx.x) * 8;
    if (i + 8 <= n) {
        float4 a = *(const float4*)(in + i);
        float4 b = *(const float4*)(in + i + 4);
        union { _Float16 h[8]; uint4v v; } u;
        u.h[0] = (_Float16)a.x; u.h[1] = (_Float16)a.y;
        u.h[2] = (_Float16)a.z; u.h[3] = (_Float16)a.w;
        u.h[4] = (_Float16)b.x; u.h[5] = (_Float16)b.y;
        u.h[6] = (_Float16)b.z; u.h[7] = (_Float16)b.w;
        *(uint4v*)(out + i) = u.v;
    }
}

// ------------- transpose + cast weights: Wt[n][k] = W[k][n] -------------
__global__ void __launch_bounds__(256)
transpose_cast_w(const float* __restrict__ Wq, const float* __restrict__ Wk,
                 const float* __restrict__ Wv,
                 _Float16* __restrict__ Tq, _Float16* __restrict__ Tk,
                 _Float16* __restrict__ Tv) {
    const float* W = (blockIdx.z == 0) ? Wq : (blockIdx.z == 1 ? Wk : Wv);
    _Float16*    T = (blockIdx.z == 0) ? Tq : (blockIdx.z == 1 ? Tk : Tv);
    int idx = blockIdx.x * 256 + threadIdx.x;      // idx = k*1024 + n
    int k = idx >> 10, n = idx & 1023;
    T[(size_t)n * DMODEL + k] = (_Float16)W[idx];
}

// ---------------- generic 128x128-tile WMMA f16 GEMM ----------------
// C[z] = A[z] (MxK row-major, lda) * Bm[z]^T (Bm is N x K row-major, ldb)
// 8 waves/block; each wave: 16 rows x 128 cols. Double-buffered LDS fed by
// global_load_async_to_lds_b128; K-loop step 32.
__global__ void __launch_bounds__(256)
wmma_gemm_kernel(const _Float16* __restrict__ Abase,
                 const _Float16* __restrict__ Bbase,
                 const float* __restrict__ bias,
                 void* __restrict__ Cbase,
                 int Kdim, int lda, int ldb, int ldc,
                 long strideA, long strideB, long strideC,
                 int mode)
{
    __shared__ _Float16 As[2][128 * 32];   // 2 x 8 KB, row-major [m][k]
    __shared__ _Float16 Bs[2][128 * 32];   // 2 x 8 KB, n-major   [n][k]

    const int z = blockIdx.z;
    const _Float16* A  = Abase + (size_t)z * strideA;
    const _Float16* Bm = Bbase + (size_t)z * strideB;

    const int rowBase = blockIdx.y * 128;
    const int colBase = blockIdx.x * 128;

    const int tid  = threadIdx.x;
    const int wave = tid >> 5;          // 0..7
    const int lane = tid & 31;
    const int half = lane >> 4;         // 0/1
    const int l16  = lane & 15;

    // staging: each thread moves 32 B of A tile + 32 B of B tile (2 b128 each)
    const int ldm = tid >> 1;           // row/col-line 0..127
    const int ldk = (tid & 1) * 16;     // half offset 0 or 16 (covers 16 halfs)

    const _Float16* gA0 = A  + (size_t)(rowBase + ldm) * lda + ldk;
    const _Float16* gB0 = Bm + (size_t)(colBase + ldm) * ldb + ldk;

    auto issue_tile = [&](int kk, int buf) {
        unsigned la = (unsigned)(size_t)(&As[buf][ldm * 32 + ldk]);
        unsigned lb = (unsigned)(size_t)(&Bs[buf][ldm * 32 + ldk]);
        async_copy_b128(la,      (unsigned long long)(size_t)(gA0 + kk));
        async_copy_b128(la + 16, (unsigned long long)(size_t)(gA0 + kk + 8));
        async_copy_b128(lb,      (unsigned long long)(size_t)(gB0 + kk));
        async_copy_b128(lb + 16, (unsigned long long)(size_t)(gB0 + kk + 8));
    };

    v8f acc[8] = {};

    const int nIter = Kdim >> 5;
    issue_tile(0, 0);

    for (int it = 0; it < nIter; ++it) {
        const int cur = it & 1;
        wait_async0();          // my async writes into buf `cur` have landed in LDS
        __syncthreads();        // everyone's writes landed; everyone's prior reads done
        if (it + 1 < nIter) issue_tile((it + 1) << 5, cur ^ 1);
        if (it + 2 < nIter) {   // global_prefetch_b8 two K-tiles ahead
            __builtin_prefetch(gA0 + ((it + 2) << 5), 0, 1);
            __builtin_prefetch(gB0 + ((it + 2) << 5), 0, 1);
        }

        // A fragment (16-bit 16x32): lanes 0-15 K=0..7,16..23 ; lanes 16-31 K=8..15,24..31
        const uint4v* ap = (const uint4v*)(&As[cur][(wave * 16 + l16) * 32]);
        union { uint4v u[2]; v16h h; } ua;
        ua.u[0] = ap[half];
        ua.u[1] = ap[2 + half];
        const v16h afrag = ua.h;

        // hoist all 8 B fragments, then stream 8 independent WMMAs
        v16h bfrag[8];
        #pragma unroll
        for (int nt = 0; nt < 8; ++nt) {
            const uint4v* bp = (const uint4v*)(&Bs[cur][(nt * 16 + l16) * 32]);
            union { uint4v u[2]; v16h h; } ub;
            ub.u[0] = bp[half];
            ub.u[1] = bp[2 + half];
            bfrag[nt] = ub.h;
        }
        #pragma unroll
        for (int nt = 0; nt < 8; ++nt) {
            acc[nt] = __builtin_amdgcn_wmma_f32_16x16x32_f16(
                false, afrag, false, bfrag[nt], (short)0, acc[nt], false, false);
        }
    }

    // epilogue: C element (vgpr j, lane) -> row = j + 8*half, col = l16
    #pragma unroll
    for (int nt = 0; nt < 8; ++nt) {
        #pragma unroll
        for (int j = 0; j < 8; ++j) {
            const int r = rowBase + wave * 16 + j + half * 8;
            const int c = colBase + nt * 16 + l16;
            float v = acc[nt][j];
            if (mode == EPI_F32) {
                float* C = (float*)Cbase + (size_t)z * strideC;
                C[(size_t)r * ldc + c] = v;
            } else if (mode == EPI_BIAS_F16) {
                v += bias[c];
                _Float16* C = (_Float16*)Cbase;
                C[(size_t)r * ldc + c] = (_Float16)v;
            } else { // EPI_BIAS_F16_TRANS: V stored as Vt[b][v_col][token]
                v += bias[c];
                const int b   = r >> 11;     // r / SEQ
                const int tok = r & (SEQ - 1);
                _Float16* C = (_Float16*)Cbase;
                C[((size_t)b * DMODEL + c) * SEQ + tok] = (_Float16)v;
            }
        }
    }
}

// -------- row softmax over SEQ=2048 f32; writes f16 P in place --------
// P row r lives in the first 4096 bytes of S row r (half stride 4096).
__global__ void __launch_bounds__(256)
softmax_rows(float* __restrict__ S) {
    const size_t row = blockIdx.x;
    float*    s = S + row * (size_t)SEQ;
    _Float16* p = (_Float16*)s;
    const int t = threadIdx.x;

    float v[8];
    float m = -1e30f;
    #pragma unroll
    for (int i = 0; i < 8; ++i) { v[i] = s[t + i * 256]; m = fmaxf(m, v[i]); }

    __shared__ float red[256];
    red[t] = m; __syncthreads();
    for (int off = 128; off > 0; off >>= 1) {
        if (t < off) red[t] = fmaxf(red[t], red[t + off]);
        __syncthreads();
    }
    m = red[0]; __syncthreads();

    float sum = 0.f;
    #pragma unroll
    for (int i = 0; i < 8; ++i) { v[i] = __expf(v[i] - m); sum += v[i]; }
    red[t] = sum; __syncthreads();
    for (int off = 128; off > 0; off >>= 1) {
        if (t < off) red[t] += red[t + off];
        __syncthreads();
    }
    const float inv = 1.f / red[0];

    #pragma unroll
    for (int i = 0; i < 8; ++i) p[t + i * 256] = (_Float16)(v[i] * inv);
}

extern "C" void kernel_launch(void* const* d_in, const int* in_sizes, int n_in,
                              void* d_out, int out_size, void* d_ws, size_t ws_size,
                              hipStream_t stream) {
    const float* x  = (const float*)d_in[0];
    const float* Wq = (const float*)d_in[1];
    const float* bq = (const float*)d_in[2];
    const float* Wk = (const float*)d_in[3];
    const float* bk = (const float*)d_in[4];
    const float* Wv = (const float*)d_in[5];
    const float* bv = (const float*)d_in[6];
    float* out = (float*)d_out;

    const size_t M = (size_t)BATCH * SEQ;                // 16384
    char* w = (char*)d_ws;
    _Float16* xh = (_Float16*)w;  w += M * DMODEL * 2;                  // 32 MB
    _Float16* Tq = (_Float16*)w;  w += (size_t)DMODEL * DMODEL * 2;     // 2 MB
    _Float16* Tk = (_Float16*)w;  w += (size_t)DMODEL * DMODEL * 2;
    _Float16* Tv = (_Float16*)w;  w += (size_t)DMODEL * DMODEL * 2;
    _Float16* Qh = (_Float16*)w;  w += M * DMODEL * 2;                  // 32 MB
    _Float16* Kh = (_Float16*)w;  w += M * DMODEL * 2;                  // 32 MB
    _Float16* Vt = (_Float16*)w;  w += M * DMODEL * 2;                  // 32 MB (transposed)
    float*    S  = (float*)w;     // 8*2048*2048*4 = 128 MB (L2-resident); P aliases it

    // 1) casts
    cast_f32_to_f16<<<8192, 256, 0, stream>>>(x, xh, (int)(M * DMODEL));
    transpose_cast_w<<<dim3(4096, 1, 3), 256, 0, stream>>>(Wq, Wk, Wv, Tq, Tk, Tv);

    // 2) QKV projections: M=16384, N=1024, K=1024
    dim3 gq(DMODEL / 128, M / 128, 1);
    wmma_gemm_kernel<<<gq, 256, 0, stream>>>(xh, Tq, bq, Qh,
        DMODEL, DMODEL, DMODEL, DMODEL, 0, 0, 0, EPI_BIAS_F16);
    wmma_gemm_kernel<<<gq, 256, 0, stream>>>(xh, Tk, bk, Kh,
        DMODEL, DMODEL, DMODEL, DMODEL, 0, 0, 0, EPI_BIAS_F16);
    wmma_gemm_kernel<<<gq, 256, 0, stream>>>(xh, Tv, bv, Vt,
        DMODEL, DMODEL, DMODEL, DMODEL, 0, 0, 0, EPI_BIAS_F16_TRANS);

    // 3) scores S = Q K^T per batch: 2048x2048, K=1024
    dim3 gs(SEQ / 128, SEQ / 128, BATCH);
    wmma_gemm_kernel<<<gs, 256, 0, stream>>>(Qh, Kh, nullptr, S,
        DMODEL, DMODEL, DMODEL, SEQ,
        (long)SEQ * DMODEL, (long)SEQ * DMODEL, (long)SEQ * SEQ, EPI_F32);

    // 4) softmax rows -> P (f16, in place, row stride 4096 halfs)
    softmax_rows<<<(unsigned)M, 256, 0, stream>>>(S);

    // 5) out = P @ V per batch: 2048x1024, K=2048 (A = P, lda = 4096 halfs)
    const _Float16* P = (const _Float16*)S;
    dim3 go(DMODEL / 128, SEQ / 128, BATCH);
    wmma_gemm_kernel<<<go, 256, 0, stream>>>(P, Vt, nullptr, out,
        SEQ, 2 * SEQ, SEQ, DMODEL,
        (long)SEQ * 2 * SEQ, (long)DMODEL * SEQ, (long)SEQ * DMODEL, EPI_F32);
}
